// GAT_43533788512794
// MI455X (gfx1250) — matrix operand order repaired and checked by reading, compiled, not run on hardware
//
#include <hip/hip_runtime.h>
#include <hip/hip_bf16.h>
#include <math.h>

// ---------------------------------------------------------------------------
// CDNA5 WMMA types
// ---------------------------------------------------------------------------
typedef __attribute__((ext_vector_type(16))) _Float16 v16h;
typedef __attribute__((ext_vector_type(8)))  float    v8f;
typedef __attribute__((ext_vector_type(4)))  float    v4f;

#define GAT_NEG_SLOPE 0.2f
#define MLP_NEG_SLOPE 0.01f

// ---------------------------------------------------------------------------
// Weight conversion: W [K, Nn] f32 row-major  ->  Wt [Nn, K] f16 row-major
// ---------------------------------------------------------------------------
__global__ void wconv_kernel(const float* __restrict__ W, _Float16* __restrict__ Wt,
                             int K, int Nn) {
    int idx = blockIdx.x * blockDim.x + threadIdx.x;
    if (idx >= K * Nn) return;
    int k = idx / Nn;
    int n = idx % Nn;
    Wt[(size_t)n * K + k] = (_Float16)W[idx];
}

// ---------------------------------------------------------------------------
// FAST WMMA GEMM for perfectly tiled shapes:
//   requires  M % 16 == 0, K % 32 == 0 (compile-time), Nout % 64 == 0
//   C[M,Nout] = A[M,K](f32) * B (as f16 Bt[Nout,K]),  no bias / activation.
// One wave per 16-row tile x 64 columns: 4 accumulators, A-fragment reused
// across 4 back-to-back v_wmma per k-step; K template param => full unroll,
// guard-free straight-line loads feeding WMMA (no phi/mov churn).
//
// Fragment layouts as loaded here (wave32, contiguous k):
//   A 16x32 f16 : m=lane&15, hi=lane>>4; half h -> K=k0+16*hi+h
//   B 32x16 f16 : N=lane&15; half h -> K=k0+16*hi+h
//   C/D 16x16 f32: N=lane&15; vgpr r -> M=r+8*hi
// ---------------------------------------------------------------------------
template <int K>
__global__ void __launch_bounds__(32)
gemm_wmma_fast_kernel(const float* __restrict__ A, const _Float16* __restrict__ Bt,
                      float* __restrict__ C, int Nout) {
    const int lane  = threadIdx.x & 31;
    const int laneM = lane & 15;
    const int hi    = lane >> 4;
    const int mbase = blockIdx.x * 16;
    const int nbase = blockIdx.y * 64;

    const float*    arow = A + (size_t)(mbase + laneM) * K;
    const _Float16* bp   = Bt + (size_t)(nbase + laneM) * K + hi * 16;

    const v8f zero = {0.f, 0.f, 0.f, 0.f, 0.f, 0.f, 0.f, 0.f};
    v8f acc[4] = {zero, zero, zero, zero};

#pragma unroll
    for (int k0 = 0; k0 < K; k0 += 32) {
        // A fragment: two 16B f32 loads per half, convert to f16
        const float* ap = arow + k0 + hi * 16;
        const v4f a0 = *(const v4f*)(ap + 0);
        const v4f a1 = *(const v4f*)(ap + 4);
        const v4f a2 = *(const v4f*)(ap + 8);
        const v4f a3 = *(const v4f*)(ap + 12);
        v16h a;
#pragma unroll
        for (int j = 0; j < 4; ++j) {
            a[j]      = (_Float16)a0[j];
            a[4 + j]  = (_Float16)a1[j];
            a[8 + j]  = (_Float16)a2[j];
            a[12 + j] = (_Float16)a3[j];
        }
        // 4 N-tiles back-to-back, A reused
#pragma unroll
        for (int nt = 0; nt < 4; ++nt) {
            const v16h b = *(const v16h*)(bp + (size_t)nt * 16 * K + k0);
            acc[nt] = __builtin_amdgcn_wmma_f32_16x16x32_f16(
                false, a, false, b, (short)0, acc[nt], false, false);
        }
    }

    // store: M = mbase + hi*8 + r, N = nbase + nt*16 + laneM
#pragma unroll
    for (int nt = 0; nt < 4; ++nt) {
        const int ncol = nbase + nt * 16 + laneM;
#pragma unroll
        for (int r = 0; r < 8; ++r) {
            C[(size_t)(mbase + hi * 8 + r) * Nout + ncol] = acc[nt][r];
        }
    }
}

// ---------------------------------------------------------------------------
// Generic WMMA GEMM (bounds-checked) for irregular shapes:
// K=16 input MLP (zero-padded K), Nout=2 readout (masked columns).
// Fused bias + optional LeakyReLU(0.01).
// ---------------------------------------------------------------------------
__global__ void __launch_bounds__(32)
gemm_wmma_kernel(const float* __restrict__ A, const _Float16* __restrict__ Bt,
                 float* __restrict__ C, const float* __restrict__ bias,
                 int M, int K, int Nout, int act) {
    const int lane  = threadIdx.x & 31;
    const int laneM = lane & 15;
    const int hi    = lane >> 4;
    const int mtile = blockIdx.x;
    const int nbase = blockIdx.y * 4;

    const int  row   = mtile * 16 + laneM;
    const bool rowOK = row < M;

    const v8f zero = {0.f, 0.f, 0.f, 0.f, 0.f, 0.f, 0.f, 0.f};
    v8f acc[4];
#pragma unroll
    for (int nt = 0; nt < 4; ++nt) acc[nt] = zero;

    for (int k0 = 0; k0 < K; k0 += 32) {
        v16h a;
        if (rowOK && (k0 + 32 <= K)) {
            const float* ap = A + (size_t)row * K + k0 + hi * 16;
            const v4f a0 = *(const v4f*)(ap + 0);
            const v4f a1 = *(const v4f*)(ap + 4);
            const v4f a2 = *(const v4f*)(ap + 8);
            const v4f a3 = *(const v4f*)(ap + 12);
#pragma unroll
            for (int j = 0; j < 4; ++j) {
                a[j]      = (_Float16)a0[j];
                a[4 + j]  = (_Float16)a1[j];
                a[8 + j]  = (_Float16)a2[j];
                a[12 + j] = (_Float16)a3[j];
            }
        } else {
#pragma unroll
            for (int j = 0; j < 8; ++j) {
                const int kA = k0 + hi * 16 + j;
                const int kB = kA + 8;
                a[j]     = (_Float16)((rowOK && kA < K) ? A[(size_t)row * K + kA] : 0.f);
                a[8 + j] = (_Float16)((rowOK && kB < K) ? A[(size_t)row * K + kB] : 0.f);
            }
        }
#pragma unroll
        for (int nt = 0; nt < 4; ++nt) {
            const int n16 = (nbase + nt) * 16;
            if (n16 >= Nout) continue;  // uniform branch, EXEC stays all-ones
            const int ncol = n16 + laneM;
            v16h b;
            if (ncol < Nout && (k0 + 32 <= K)) {
                b = *(const v16h*)(Bt + (size_t)ncol * K + k0 + hi * 16);
            } else {
#pragma unroll
                for (int j = 0; j < 16; ++j) {
                    const int kk = k0 + hi * 16 + j;
                    b[j] = (ncol < Nout && kk < K) ? Bt[(size_t)ncol * K + kk]
                                                   : (_Float16)0.f;
                }
            }
            acc[nt] = __builtin_amdgcn_wmma_f32_16x16x32_f16(
                false, a, false, b, (short)0, acc[nt], false, false);
        }
    }

#pragma unroll
    for (int nt = 0; nt < 4; ++nt) {
        const int n16 = (nbase + nt) * 16;
        if (n16 >= Nout) continue;
        const int ncol = n16 + laneM;
        const float bv = (bias != nullptr && ncol < Nout) ? bias[ncol] : 0.f;
#pragma unroll
        for (int r = 0; r < 8; ++r) {
            const int mr = mtile * 16 + hi * 8 + r;
            float v = acc[nt][r] + bv;
            if (act) v = (v > 0.f) ? v : MLP_NEG_SLOPE * v;
            if (mr < M && ncol < Nout) C[(size_t)mr * Nout + ncol] = v;
        }
    }
}

// ---------------------------------------------------------------------------
// Attention scores: a_src[n,h] = sum_c xl[n,h,c]*att_src[h,c]  (and a_dst)
// ---------------------------------------------------------------------------
__global__ void attn_kernel(const float* __restrict__ xl,
                            const float* __restrict__ att_s,
                            const float* __restrict__ att_d,
                            float* __restrict__ aS, float* __restrict__ aD,
                            int N, int heads, int oc) {
    int t = blockIdx.x * blockDim.x + threadIdx.x;
    if (t >= N * heads) return;
    int n = t / heads;
    int h = t % heads;
    const float* xp = xl + (size_t)n * heads * oc + (size_t)h * oc;
    const float* as = att_s + h * oc;
    const float* ad = att_d + h * oc;
    float s = 0.f, d = 0.f;
    for (int c = 0; c < oc; ++c) {
        float v = xp[c];
        s += v * as[c];
        d += v * ad[c];
    }
    aS[t] = s;
    aD[t] = d;
}

// ---------------------------------------------------------------------------
// Small utility kernels
// ---------------------------------------------------------------------------
__global__ void fill_i32_kernel(int* __restrict__ p, int v, int n) {
    int i = blockIdx.x * blockDim.x + threadIdx.x;
    if (i < n) p[i] = v;
}
__global__ void fill_f32_kernel(float* __restrict__ p, float v, int n) {
    int i = blockIdx.x * blockDim.x + threadIdx.x;
    if (i < n) p[i] = v;
}
__global__ void bias_bcast_kernel(float* __restrict__ out, const float* __restrict__ bias,
                                  int total, int H) {
    int i = blockIdx.x * blockDim.x + threadIdx.x;
    if (i < total) out[i] = bias[i % H];
}
__global__ void sub_kernel(const float* __restrict__ a, const float* __restrict__ b,
                           float* __restrict__ o, int n) {
    int i = blockIdx.x * blockDim.x + threadIdx.x;
    if (i < n) o[i] = a[i] - b[i];
}
__global__ void mul_kernel(const float* __restrict__ a, const float* __restrict__ b,
                           float* __restrict__ o, int n) {
    int i = blockIdx.x * blockDim.x + threadIdx.x;
    if (i < n) o[i] = a[i] * b[i];
}

// ---------------------------------------------------------------------------
// Monotonic float<->int encoding so float max can use integer atomicMax
// ---------------------------------------------------------------------------
__device__ __forceinline__ int enc_f(float v) {
    int b = __float_as_int(v);
    return (b >= 0) ? b : (b ^ 0x7fffffff);
}
__device__ __forceinline__ float dec_f(int b) {
    return __int_as_float((b >= 0) ? b : (b ^ 0x7fffffff));
}

// edges e < E come from edge_index; e >= E are implicit self-loops (node e-E)
__global__ void edge_max_kernel(const int* __restrict__ src, const int* __restrict__ dst,
                                int E, int Etot,
                                const float* __restrict__ aS, const float* __restrict__ aD,
                                int* __restrict__ amax, int heads) {
    int e = blockIdx.x * blockDim.x + threadIdx.x;
    if (e >= Etot) return;
    int s = (e < E) ? src[e] : (e - E);
    int d = (e < E) ? dst[e] : (e - E);
    for (int h = 0; h < heads; ++h) {
        float a = aS[(size_t)s * heads + h] + aD[(size_t)d * heads + h];
        a = (a > 0.f) ? a : GAT_NEG_SLOPE * a;
        atomicMax(&amax[(size_t)d * heads + h], enc_f(a));
    }
}

__global__ void edge_sum_kernel(const int* __restrict__ src, const int* __restrict__ dst,
                                int E, int Etot,
                                const float* __restrict__ aS, const float* __restrict__ aD,
                                const int* __restrict__ amax,
                                float* __restrict__ exbuf, float* __restrict__ denom,
                                int heads) {
    int e = blockIdx.x * blockDim.x + threadIdx.x;
    if (e >= Etot) return;
    int s = (e < E) ? src[e] : (e - E);
    int d = (e < E) ? dst[e] : (e - E);
    for (int h = 0; h < heads; ++h) {
        float a = aS[(size_t)s * heads + h] + aD[(size_t)d * heads + h];
        a = (a > 0.f) ? a : GAT_NEG_SLOPE * a;
        float ex = expf(a - dec_f(amax[(size_t)d * heads + h]));
        exbuf[(size_t)e * heads + h] = ex;
        atomicAdd(&denom[(size_t)d * heads + h], ex);
    }
}

// block per edge, thread per channel (H = heads*oc = 128)
__global__ void edge_scatter_kernel(const int* __restrict__ src, const int* __restrict__ dst,
                                    int E, int Etot,
                                    const float* __restrict__ exbuf,
                                    const float* __restrict__ denom,
                                    const float* __restrict__ xl, float* __restrict__ agg,
                                    int heads, int oc) {
    int e = blockIdx.x;
    if (e >= Etot) return;
    int i = threadIdx.x;  // 0..heads*oc-1
    int s = (e < E) ? src[e] : (e - E);
    int d = (e < E) ? dst[e] : (e - E);
    int h = i / oc;
    int H = heads * oc;
    float coef = exbuf[(size_t)e * heads + h] /
                 (denom[(size_t)d * heads + h] + 1e-16f);
    atomicAdd(&agg[(size_t)d * H + i], xl[(size_t)s * H + i] * coef);
}

// ---------------------------------------------------------------------------
// Host-side orchestration
// ---------------------------------------------------------------------------
extern "C" void kernel_launch(void* const* d_in, const int* in_sizes, int n_in,
                              void* d_out, int out_size, void* d_ws, size_t ws_size,
                              hipStream_t stream) {
    const float* mask_feature = (const float*)d_in[0];
    const float* feature      = (const float*)d_in[1];
    const int*   edge_index   = (const int*)d_in[2];
    // d_in[3] edge_type: unused by the reference
    const float* W_in     = (const float*)d_in[4];
    const float* b_in     = (const float*)d_in[5];
    const float* W1       = (const float*)d_in[6];
    const float* att_src1 = (const float*)d_in[7];
    const float* att_dst1 = (const float*)d_in[8];
    const float* b1       = (const float*)d_in[9];
    const float* W2       = (const float*)d_in[10];
    const float* att_src2 = (const float*)d_in[11];
    const float* att_dst2 = (const float*)d_in[12];
    const float* b2       = (const float*)d_in[13];
    const float* W_out    = (const float*)d_in[14];
    const float* b_out    = (const float*)d_in[15];

    const int H    = in_sizes[5];            // 128
    const int DIN  = in_sizes[4] / H;        // 16
    const int N    = in_sizes[0] / DIN;      // 100000
    const int E    = in_sizes[2] / 2;        // 600000
    const int OUT  = in_sizes[15];           // 2
    const int Etot = E + N;                  // edges + self loops
    const int H1 = 4, OC1 = H / 4;           // layer-1: 4 heads x 32
    const int H2 = 1, OC2 = H;               // layer-2: 1 head  x 128

    const int* srcIdx = edge_index;
    const int* dstIdx = edge_index + E;

    // ---- carve up workspace (256B aligned chunks) ----
    char* p = (char*)d_ws;
    auto take = [&](size_t bytes) -> void* {
        void* r = (void*)p;
        p += (bytes + 255) & ~(size_t)255;
        return r;
    };
    float*    B0    = (float*)take((size_t)N * H * 4);        // x / agg ping
    float*    B1    = (float*)take((size_t)N * H * 4);        // xl / pong
    float*    exbuf = (float*)take((size_t)Etot * H1 * 4);    // per-edge exp
    float*    aS    = (float*)take((size_t)N * H1 * 4);
    float*    aD    = (float*)take((size_t)N * H1 * 4);
    int*      amax  = (int*)take((size_t)N * H1 * 4);
    float*    denom = (float*)take((size_t)N * H1 * 4);
    float*    diffb = (float*)take((size_t)N * DIN * 4);
    float*    xfin  = (float*)take((size_t)N * OUT * 4);
    float*    mfin  = (float*)take((size_t)N * OUT * 4);
    _Float16* WtIn  = (_Float16*)take((size_t)DIN * H * 2);
    _Float16* Wt1   = (_Float16*)take((size_t)H * H * 2);
    _Float16* Wt2   = (_Float16*)take((size_t)H * H * 2);
    _Float16* WtOut = (_Float16*)take((size_t)H * OUT * 2);
    (void)ws_size; (void)n_in; (void)out_size;

    auto cdiv = [](int a, int b) { return (a + b - 1) / b; };
    // generic (guarded) GEMM for irregular shapes
    auto gemm = [&](const float* A, const _Float16* Bt, float* C, const float* bias,
                    int M, int K, int Nout, int act) {
        dim3 grid(cdiv(M, 16), cdiv(Nout, 64));
        gemm_wmma_kernel<<<grid, 32, 0, stream>>>(A, Bt, C, bias, M, K, Nout, act);
    };
    // fast GEMM: M%16==0, K==128, Nout%64==0, no bias/act
    auto gemm128 = [&](const float* A, const _Float16* Bt, float* C, int M, int Nout) {
        dim3 grid(M / 16, Nout / 64);
        gemm_wmma_fast_kernel<128><<<grid, 32, 0, stream>>>(A, Bt, C, Nout);
    };

    // ---- 0. convert weights to transposed f16 ----
    wconv_kernel<<<cdiv(DIN * H, 256), 256, 0, stream>>>(W_in, WtIn, DIN, H);
    wconv_kernel<<<cdiv(H * H, 256), 256, 0, stream>>>(W1, Wt1, H, H);
    wconv_kernel<<<cdiv(H * H, 256), 256, 0, stream>>>(W2, Wt2, H, H);
    wconv_kernel<<<cdiv(H * OUT, 256), 256, 0, stream>>>(W_out, WtOut, H, OUT);

    // ---- 1. input MLP: x0 = leaky_relu(mask_feature @ W_in + b_in) ----
    gemm(mask_feature, WtIn, B0, b_in, N, DIN, H, 1);

    // ---- 2. GAT layer 1 (4 heads x 32) ----
    if (H == 128 && (N % 16) == 0 && (H % 64) == 0)
        gemm128(B0, Wt1, B1, N, H);          // xl1 = x0 @ W1 (fast path)
    else
        gemm(B0, Wt1, B1, nullptr, N, H, H, 0);
    attn_kernel<<<cdiv(N * H1, 256), 256, 0, stream>>>(B1, att_src1, att_dst1,
                                                       aS, aD, N, H1, OC1);
    fill_i32_kernel<<<cdiv(N * H1, 256), 256, 0, stream>>>(amax, (int)0x80000000, N * H1);
    fill_f32_kernel<<<cdiv(N * H1, 256), 256, 0, stream>>>(denom, 0.f, N * H1);
    bias_bcast_kernel<<<cdiv(N * H, 256), 256, 0, stream>>>(B0, b1, N * H, H);  // agg = bias
    edge_max_kernel<<<cdiv(Etot, 256), 256, 0, stream>>>(srcIdx, dstIdx, E, Etot,
                                                         aS, aD, amax, H1);
    edge_sum_kernel<<<cdiv(Etot, 256), 256, 0, stream>>>(srcIdx, dstIdx, E, Etot,
                                                         aS, aD, amax, exbuf, denom, H1);
    edge_scatter_kernel<<<Etot, H, 0, stream>>>(srcIdx, dstIdx, E, Etot,
                                                exbuf, denom, B1, B0, H1, OC1);

    // ---- 3. GAT layer 2 (1 head x 128) ----
    if (H == 128 && (N % 16) == 0 && (H % 64) == 0)
        gemm128(B0, Wt2, B1, N, H);          // xl2 = x1 @ W2 (fast path)
    else
        gemm(B0, Wt2, B1, nullptr, N, H, H, 0);
    attn_kernel<<<cdiv(N * H2, 256), 256, 0, stream>>>(B1, att_src2, att_dst2,
                                                       aS, aD, N, H2, OC2);
    fill_i32_kernel<<<cdiv(N * H2, 256), 256, 0, stream>>>(amax, (int)0x80000000, N * H2);
    fill_f32_kernel<<<cdiv(N * H2, 256), 256, 0, stream>>>(denom, 0.f, N * H2);
    bias_bcast_kernel<<<cdiv(N * H, 256), 256, 0, stream>>>(B0, b2, N * H, H);
    edge_max_kernel<<<cdiv(Etot, 256), 256, 0, stream>>>(srcIdx, dstIdx, E, Etot,
                                                         aS, aD, amax, H2);
    edge_sum_kernel<<<cdiv(Etot, 256), 256, 0, stream>>>(srcIdx, dstIdx, E, Etot,
                                                         aS, aD, amax, exbuf, denom, H2);
    edge_scatter_kernel<<<Etot, H, 0, stream>>>(srcIdx, dstIdx, E, Etot,
                                                exbuf, denom, B1, B0, H2, OC2);

    // ---- 4. readout: xfin = x2 @ W_out + b_out ----
    gemm(B0, WtOut, xfin, b_out, N, H, OUT, 0);

    // ---- 5. mask branch ----
    sub_kernel<<<cdiv(N * DIN, 256), 256, 0, stream>>>(feature, mask_feature, diffb, N * DIN);
    gemm(diffb, WtIn, B1, b_in, N, DIN, H, 1);   // leaky MLP
    gemm(B1, WtOut, mfin, b_out, N, H, OUT, 0);  // mask projection

    // ---- 6. out = xfin * mfin ----
    mul_kernel<<<cdiv(N * OUT, 256), 256, 0, stream>>>(xfin, mfin, (float*)d_out, N * OUT);
}
